// WaveRNN_2353642078816
// MI455X (gfx1250) — compile-verified, matrix operand order checked by reading, not run on hardware
//
#include <hip/hip_runtime.h>
#include <stdint.h>

typedef __bf16        v16bf __attribute__((ext_vector_type(16)));
typedef float         v8f   __attribute__((ext_vector_type(8)));
typedef unsigned int  v4u   __attribute__((ext_vector_type(4)));
typedef int           v8i   __attribute__((ext_vector_type(8)));
typedef int           v4i   __attribute__((ext_vector_type(4)));

#define DEVI __device__ __forceinline__

// ---------------------------------------------------------------------------
// WMMA helpers (CDNA5: V_WMMA_F32_16X16X32_BF16, wave32)
// ---------------------------------------------------------------------------
DEVI v8f wmma_bf16(v16bf a, v16bf b, v8f c) {
  return __builtin_amdgcn_wmma_f32_16x16x32_bf16(false, a, false, b, (short)0, c,
                                                 false, false);
}

// B fragment: pre-packed so lane's 16 halves are contiguous (32B load).
DEVI v16bf load_bfrag(const __bf16* Bp, int NT, int kt, int nt, int lane) {
  return *(const v16bf*)(Bp + ((((size_t)kt * NT + nt) * 32 + (size_t)lane) << 4));
}

// A fragment from a row-major f32 tile (row stride 32): lane m = lane&15,
// halves 0..7 -> K = ks*8+i, halves 8..15 -> K = 16+ks*8+i  (ks = lane>>4)
DEVI v16bf afrag_from_f32(const float* p) {
  v16bf a;
#pragma unroll
  for (int i = 0; i < 8; ++i) { a[i] = (__bf16)p[i]; a[i + 8] = (__bf16)p[16 + i]; }
  return a;
}

DEVI v16bf afrag_from_bf16(const __bf16* p) {
  v16bf a;
#pragma unroll
  for (int i = 0; i < 8; ++i) { a[i] = p[i]; a[i + 8] = p[16 + i]; }
  return a;
}

// ---------------------------------------------------------------------------
// Tensor Data Mover: 2D f32 tile (tileW x tileH, row stride rowElems) -> LDS.
// D# built per CDNA5 ISA (count=1, type=2, data_size=4B).
// This toolchain exposes the 6-arg builtin (extra int32x8 group, then cpol).
// ---------------------------------------------------------------------------
DEVI void tdm_load_f32_tile(const float* gsrc, unsigned ldsOff,
                            unsigned rowElems, unsigned tileW, unsigned tileH) {
  unsigned long long ga = (unsigned long long)(uintptr_t)gsrc;
  v4u g0;
  g0.x = 1u;                                             // count=1, is_restore=0
  g0.y = ldsOff;                                         // lds_addr
  g0.z = (unsigned)ga;                                   // global_addr[31:0]
  g0.w = ((unsigned)(ga >> 32) & 0x01FFFFFFu) | 0x80000000u;  // ga[56:32] | type=2
  v8i g1;
  g1[0] = (int)(2u << 16);                               // data_size = 4B
  g1[1] = (int)((rowElems & 0xFFFFu) << 16);             // tensor_dim0[15:0]
  g1[2] = (int)((rowElems >> 16) & 0xFFFFu);             // dim0[31:16] | dim1[15:0]=0
  g1[3] = (int)(0x10u | ((tileW & 0xFFFFu) << 16));      // dim1=0x100000, tile_dim0
  g1[4] = (int)(tileH & 0xFFFFu);                        // tile_dim1, tile_dim2=0
  g1[5] = (int)rowElems;                                 // tensor_dim0_stride[31:0]
  g1[6] = 0;
  g1[7] = 0;
  v4i z4 = {0, 0, 0, 0};
  v8i z8 = {0, 0, 0, 0, 0, 0, 0, 0};
  __builtin_amdgcn_tensor_load_to_lds(g0, g1, z4, z4, z8, 0);
}

// ---------------------------------------------------------------------------
// Big batched GEMM:  C[M,N] = A[M,K](f32) * Bp(bf16 packed, N x K)^T + bias
// block: 64(M) x 128(N), 8 waves (2x4), each wave 2x2 16x16 WMMA accs.
// A tiles double-buffered through LDS via TDM.
// ---------------------------------------------------------------------------
__global__ __launch_bounds__(256) void gemm_bf16(
    const float* __restrict__ A, const __bf16* __restrict__ Bp,
    const float* __restrict__ bias, float* __restrict__ Cout,
    int Mreal, int K, int N, int doRelu) {
  __shared__ float tileA[2][64][32];
  const int tid  = threadIdx.x;
  const int lane = tid & 31;
  const int wave = tid >> 5;
  const int wm   = wave >> 2;              // 0..1
  const int wn   = wave & 3;               // 0..3
  const int bm   = blockIdx.y * 64;
  const int bn   = blockIdx.x * 128;
  const int NT   = N >> 4;
  const int KT   = K >> 5;
  const int nt0  = (bn >> 4) + wn * 2;
  const int nt1  = nt0 + 1;

  v8f acc00 = {}, acc01 = {}, acc10 = {}, acc11 = {};

  if (wave == 0)
    tdm_load_f32_tile(A + (size_t)bm * K, (unsigned)(uintptr_t)&tileA[0][0][0],
                      (unsigned)K, 32u, 64u);

  for (int kt = 0; kt < KT; ++kt) {
    const int buf = kt & 1;
    if (wave == 0) {
      if (kt + 1 < KT) {
        tdm_load_f32_tile(A + (size_t)bm * K + (size_t)(kt + 1) * 32,
                          (unsigned)(uintptr_t)&tileA[buf ^ 1][0][0],
                          (unsigned)K, 32u, 64u);
        __builtin_amdgcn_s_wait_tensorcnt((short)1);     // previous tile done
      } else {
        __builtin_amdgcn_s_wait_tensorcnt((short)0);
      }
    }
    __syncthreads();

    if (kt + 1 < KT) {                                   // L2 prefetch next B frags
      __builtin_prefetch(Bp + ((((size_t)(kt + 1) * NT + nt0) * 32 + lane) << 4), 0, 0);
      __builtin_prefetch(Bp + ((((size_t)(kt + 1) * NT + nt1) * 32 + lane) << 4), 0, 0);
    }
    v16bf a0 = afrag_from_f32(&tileA[buf][wm * 32 + (lane & 15)][(lane >> 4) << 3]);
    v16bf a1 = afrag_from_f32(&tileA[buf][wm * 32 + 16 + (lane & 15)][(lane >> 4) << 3]);
    v16bf b0 = load_bfrag(Bp, NT, kt, nt0, lane);
    v16bf b1 = load_bfrag(Bp, NT, kt, nt1, lane);
    acc00 = wmma_bf16(a0, b0, acc00);
    acc01 = wmma_bf16(a0, b1, acc01);
    acc10 = wmma_bf16(a1, b0, acc10);
    acc11 = wmma_bf16(a1, b1, acc11);
    __syncthreads();
  }

  // D layout: VGPR r -> M=r (lanes 0-15) / M=r+8 (lanes 16-31), N = lane&15
  const int col0 = (nt0 << 4) + (lane & 15);
  const int col1 = (nt1 << 4) + (lane & 15);
  const int rsel = (lane >> 4) << 3;
  const float bia0 = bias ? bias[col0] : 0.f;
  const float bia1 = bias ? bias[col1] : 0.f;
#pragma unroll
  for (int r = 0; r < 8; ++r) {
    int row0 = bm + wm * 32 + rsel + r;
    int row1 = row0 + 16;
    float v;
    v = acc00[r] + bia0; if (doRelu) v = fmaxf(v, 0.f);
    if (row0 < Mreal) Cout[(size_t)row0 * N + col0] = v;
    v = acc01[r] + bia1; if (doRelu) v = fmaxf(v, 0.f);
    if (row0 < Mreal) Cout[(size_t)row0 * N + col1] = v;
    v = acc10[r] + bia0; if (doRelu) v = fmaxf(v, 0.f);
    if (row1 < Mreal) Cout[(size_t)row1 * N + col0] = v;
    v = acc11[r] + bia1; if (doRelu) v = fmaxf(v, 0.f);
    if (row1 < Mreal) Cout[(size_t)row1 * N + col1] = v;
  }
}

// ---------------------------------------------------------------------------
// GRU scan: persistent single workgroup (32 waves), h kept in LDS as bf16
// (16x512, batch rows 8..15 zero). Per step: gh = h @ Whh^T via WMMA
// (96 N-tiles / 32 waves = 3 each, 16 K-steps), then gate math.
// xp already holds x@Wih^T + bih.  gout[row=b*T+t][j] = h_t.
// ---------------------------------------------------------------------------
__global__ __launch_bounds__(1024) void gru_scan(
    const float* __restrict__ xp, const __bf16* __restrict__ Wp,
    const float* __restrict__ bhh, float* __restrict__ gout, int T) {
  __shared__ __bf16 hA[16][512];     // 16 KB WMMA A operand
  __shared__ float  gh[8][1536];     // 48 KB gate pre-activations
  const int tid  = threadIdx.x;
  const int lane = tid & 31;
  const int wave = tid >> 5;

  for (int i = tid; i < 16 * 512; i += 1024) ((__bf16*)hA)[i] = (__bf16)0.f;
  __syncthreads();

  for (int t = 0; t < T; ++t) {
    for (int q = 0; q < 3; ++q) {
      const int nt = wave * 3 + q;
      v8f acc = {};
#pragma unroll
      for (int kt = 0; kt < 16; ++kt) {
        v16bf a = afrag_from_bf16(&hA[lane & 15][(kt << 5) + ((lane >> 4) << 3)]);
        v16bf b = load_bfrag(Wp, 96, kt, nt, lane);
        __builtin_prefetch(Wp + ((((size_t)(kt + 1) * 96 + nt) * 32 + lane) << 4), 0, 0);
        acc = wmma_bf16(a, b, acc);
      }
      if (lane < 16) {                     // lanes 0-15 hold batch rows 0..7
        const float bb = bhh[(nt << 4) + lane];
#pragma unroll
        for (int r = 0; r < 8; ++r) gh[r][(nt << 4) + lane] = acc[r] + bb;
      }
    }
    __syncthreads();

#pragma unroll
    for (int p = 0; p < 4; ++p) {          // 4096 gate units / 1024 threads
      const int idx = (p << 10) + tid;
      const int b   = idx >> 9;
      const int j   = idx & 511;
      const size_t row = (size_t)b * T + t;
      const float* xr = xp + row * 1536;
      const float rg = 1.f / (1.f + __expf(-(xr[j] + gh[b][j])));
      const float zg = 1.f / (1.f + __expf(-(xr[512 + j] + gh[b][512 + j])));
      const float ng = tanhf(xr[1024 + j] + rg * gh[b][1024 + j]);
      const float ho = (float)hA[b][j];
      const float hn = (1.f - zg) * ng + zg * ho;
      gout[row * 512 + j] = hn;
      hA[b][j] = (__bf16)hn;
    }
    __syncthreads();
  }
}

// ---------------------------------------------------------------------------
// Weight pre-pack: W (N x Ksrc, row-major f32) -> per-fragment bf16 layout.
// ---------------------------------------------------------------------------
__global__ void pack_w(const float* __restrict__ W, __bf16* __restrict__ out,
                       int Ksrc, int Kpad, int N, int total) {
  int id = blockIdx.x * blockDim.x + threadIdx.x;
  if (id >= total) return;
  const int h    = id & 15;
  const int lane = (id >> 4) & 31;
  const int frag = id >> 9;
  const int NT   = N >> 4;
  const int kt   = frag / NT;
  const int nt   = frag - kt * NT;
  const int n    = (nt << 4) + (lane & 15);
  const int ks   = lane >> 4;
  const int k    = (kt << 5) + (h < 8 ? ks * 8 + h : 16 + ks * 8 + (h - 8));
  float v = (k < Ksrc) ? W[(size_t)n * Ksrc + k] : 0.f;
  out[id] = (__bf16)v;
}

// ---------------------------------------------------------------------------
// Front-end kernels
// ---------------------------------------------------------------------------
__global__ void upsample_stage(const float* __restrict__ in, const float* __restrict__ w,
                               float* __restrict__ out, int Win, int s, int L,
                               int Wout, int total) {
  int id = blockIdx.x * blockDim.x + threadIdx.x;
  if (id >= total) return;
  const int i  = id % Wout;
  const int bf = id / Wout;
  const float* src = in + (size_t)bf * Win;
  const int rlen = Win * s;
  float acc = 0.f;
  for (int k = 0; k < L; ++k) {
    int j = i + k - s;
    if (j >= 0 && j < rlen) acc += w[k] * src[j / s];
  }
  out[id] = acc;
}

__global__ void conv_in_bn_relu(const float* __restrict__ mels,
                                const float* __restrict__ cw,
                                const float* __restrict__ bn,
                                float* __restrict__ out) {
  int id = blockIdx.x * blockDim.x + threadIdx.x;
  if (id >= 8 * 128 * 28) return;
  const int w = id % 28;
  const int o = (id / 28) % 128;
  const int b = id / (28 * 128);
  float acc = 0.f;
  for (int f = 0; f < 80; ++f) {
    const float* mp = mels + ((size_t)b * 80 + f) * 32 + w;
    const float* cp = cw + ((size_t)o * 80 + f) * 5;
#pragma unroll
    for (int kk = 0; kk < 5; ++kk) acc += mp[kk] * cp[kk];
  }
  const float sc = bn[o] * rsqrtf(bn[384 + o] + 1e-5f);
  float v = (acc - bn[256 + o]) * sc + bn[128 + o];
  out[id] = fmaxf(v, 0.f);
}

__global__ void pw_conv(const float* __restrict__ x, const float* __restrict__ wt,
                        const float* __restrict__ bn, const float* __restrict__ bias,
                        const float* __restrict__ resid, float* __restrict__ out,
                        int doRelu) {
  int id = blockIdx.x * blockDim.x + threadIdx.x;
  if (id >= 8 * 128 * 28) return;
  const int w = id % 28;
  const int o = (id / 28) % 128;
  const int b = id / (28 * 128);
  const float* xin = x + (size_t)b * 128 * 28 + w;
  const float* wr  = wt + (size_t)o * 128;
  float acc = 0.f;
  for (int c = 0; c < 128; ++c) acc += xin[(size_t)c * 28] * wr[c];
  if (bn) {
    const float sc = bn[o] * rsqrtf(bn[384 + o] + 1e-5f);
    acc = (acc - bn[256 + o]) * sc + bn[128 + o];
  }
  if (bias)  acc += bias[o];
  if (resid) acc += resid[id];
  if (doRelu) acc = fmaxf(acc, 0.f);
  out[id] = acc;
}

__global__ void build_x0(const float* __restrict__ x, const float* __restrict__ m2,
                         const float* __restrict__ aux, float* __restrict__ X0,
                         int Mp) {
  int id = blockIdx.x * blockDim.x + threadIdx.x;
  if (id >= Mp * 128) return;
  const int col = id & 127;
  const int row = id >> 7;
  float v = 0.f;
  if (row < 61600) {
    const int b = row / 7700;
    const int t = row - b * 7700;
    if (col == 0)       v = x[(size_t)b * 7700 + t];
    else if (col < 81)  v = m2[((size_t)b * 80 + (col - 1)) * 8800 + 550 + t];
    else if (col < 113) v = aux[((size_t)b * 128 + (col - 81)) * 28 + t / 275];
  }
  X0[id] = v;
}

__global__ void build_cat(const float* __restrict__ mainp, int mainStride,
                          const float* __restrict__ residp, int residStride,
                          const float* __restrict__ aux, int auxofs,
                          float* __restrict__ out, int Mp) {
  long long id = (long long)blockIdx.x * blockDim.x + threadIdx.x;
  if (id >= (long long)Mp * 544) return;
  const int c   = (int)(id % 544);
  const int row = (int)(id / 544);
  float v = 0.f;
  if (row < 61600) {
    if (c < 512) {
      v = mainp[(size_t)row * mainStride + c];
      if (residp) v += residp[(size_t)row * residStride + c];
    } else {
      const int b = row / 7700;
      const int t = row - b * 7700;
      v = aux[((size_t)b * 128 + auxofs + (c - 512)) * 28 + t / 275];
    }
  }
  out[id] = v;
}

// ---------------------------------------------------------------------------
// Host orchestration
// ---------------------------------------------------------------------------
extern "C" void kernel_launch(void* const* d_in, const int* in_sizes, int n_in,
                              void* d_out, int out_size, void* d_ws, size_t ws_size,
                              hipStream_t stream) {
  (void)in_sizes; (void)n_in; (void)out_size; (void)ws_size;
  const float* x      = (const float*)d_in[0];
  const float* mels   = (const float*)d_in[1];
  const float* cinw   = (const float*)d_in[2];
  const float* bn_in  = (const float*)d_in[3];
  const float* res1w  = (const float*)d_in[4];
  const float* rbn1   = (const float*)d_in[5];
  const float* res2w  = (const float*)d_in[6];
  const float* rbn2   = (const float*)d_in[7];
  const float* cow    = (const float*)d_in[8];
  const float* cob    = (const float*)d_in[9];
  const float* up0    = (const float*)d_in[10];
  const float* up1    = (const float*)d_in[11];
  const float* up2    = (const float*)d_in[12];
  const float* I_w    = (const float*)d_in[13];
  const float* I_b    = (const float*)d_in[14];
  const float* wih1   = (const float*)d_in[15];
  const float* whh1   = (const float*)d_in[16];
  const float* bih1   = (const float*)d_in[17];
  const float* bhh1   = (const float*)d_in[18];
  const float* wih2   = (const float*)d_in[19];
  const float* whh2   = (const float*)d_in[20];
  const float* bih2   = (const float*)d_in[21];
  const float* bhh2   = (const float*)d_in[22];
  const float* fc1w   = (const float*)d_in[23];
  const float* fc1b   = (const float*)d_in[24];
  const float* fc2w   = (const float*)d_in[25];
  const float* fc2b   = (const float*)d_in[26];
  const float* fc3w   = (const float*)d_in[27];
  const float* fc3b   = (const float*)d_in[28];

  const int T = 7700, Mreal = 61600, Mp = 61632;    // Mp = 963*64

  char* ws = (char*)d_ws;
  size_t off = 0;
  auto carve = [&](size_t bytes) {
    char* p = ws + off;
    off += (bytes + 255) & ~(size_t)255;
    return p;
  };
  float* m0   = (float*)carve((size_t)8 * 80 * 160 * 4);
  float* m1   = (float*)carve((size_t)8 * 80 * 800 * 4);
  float* m2   = (float*)carve((size_t)8 * 80 * 8800 * 4);
  float* hb0  = (float*)carve((size_t)8 * 128 * 28 * 4);
  float* hb1  = (float*)carve((size_t)8 * 128 * 28 * 4);
  float* auxb = (float*)carve((size_t)8 * 128 * 28 * 4);
  float* X0   = (float*)carve((size_t)Mp * 128 * 4);
  float* xi   = (float*)carve((size_t)Mp * 512 * 4);   // xi -> g2 -> f2
  float* xp   = (float*)carve((size_t)Mp * 1536 * 4);  // xp1 then xp2
  float* g1   = (float*)carve((size_t)Mp * 512 * 4);   // g1 then f1
  float* X2   = (float*)carve((size_t)Mp * 544 * 4);   // X2/xi1 then X4
  float* X3   = (float*)carve((size_t)Mp * 544 * 4);
  __bf16* Iwp   = (__bf16*)carve((size_t)128 * 512 * 2);
  __bf16* wih1p = (__bf16*)carve((size_t)512 * 1536 * 2);
  __bf16* whh1p = (__bf16*)carve((size_t)512 * 1536 * 2);
  __bf16* wih2p = (__bf16*)carve((size_t)544 * 1536 * 2);
  __bf16* whh2p = (__bf16*)carve((size_t)512 * 1536 * 2);
  __bf16* fc1p  = (__bf16*)carve((size_t)544 * 512 * 2);
  __bf16* fc2p  = (__bf16*)carve((size_t)544 * 512 * 2);
  __bf16* fc3p  = (__bf16*)carve((size_t)512 * 512 * 2);

  const int TB = 256;
  auto nb = [](long long n, int tb) { return (unsigned)((n + tb - 1) / tb); };

  // --- mel upsample (repeat+box-conv fused per stage) ---
  upsample_stage<<<nb(8LL * 80 * 160, TB), TB, 0, stream>>>(mels, up0, m0, 32, 5, 11, 160, 8 * 80 * 160);
  upsample_stage<<<nb(8LL * 80 * 800, TB), TB, 0, stream>>>(m0, up1, m1, 160, 5, 11, 800, 8 * 80 * 800);
  upsample_stage<<<nb(8LL * 80 * 8800, TB), TB, 0, stream>>>(m1, up2, m2, 800, 11, 23, 8800, 8 * 80 * 8800);

  // --- conv stack -> aux ---
  conv_in_bn_relu<<<nb(8 * 128 * 28, TB), TB, 0, stream>>>(mels, cinw, bn_in, hb0);
  for (int i = 0; i < 10; ++i) {
    pw_conv<<<nb(8 * 128 * 28, TB), TB, 0, stream>>>(hb0, res1w + (size_t)i * 128 * 128,
        rbn1 + (size_t)i * 4 * 128, nullptr, nullptr, hb1, 1);
    pw_conv<<<nb(8 * 128 * 28, TB), TB, 0, stream>>>(hb1, res2w + (size_t)i * 128 * 128,
        rbn2 + (size_t)i * 4 * 128, nullptr, hb0, hb0, 0);
  }
  pw_conv<<<nb(8 * 128 * 28, TB), TB, 0, stream>>>(hb0, cow, nullptr, cob, nullptr, auxb, 0);

  // --- weight pre-pack to WMMA bf16 fragment layout ---
  pack_w<<<nb(128LL * 512, TB), TB, 0, stream>>>(I_w, Iwp, 113, 128, 512, 128 * 512);
  pack_w<<<nb(512LL * 1536, TB), TB, 0, stream>>>(wih1, wih1p, 512, 512, 1536, 512 * 1536);
  pack_w<<<nb(512LL * 1536, TB), TB, 0, stream>>>(whh1, whh1p, 512, 512, 1536, 512 * 1536);
  pack_w<<<nb(544LL * 1536, TB), TB, 0, stream>>>(wih2, wih2p, 544, 544, 1536, 544 * 1536);
  pack_w<<<nb(512LL * 1536, TB), TB, 0, stream>>>(whh2, whh2p, 512, 512, 1536, 512 * 1536);
  pack_w<<<nb(544LL * 512, TB), TB, 0, stream>>>(fc1w, fc1p, 544, 544, 512, 544 * 512);
  pack_w<<<nb(544LL * 512, TB), TB, 0, stream>>>(fc2w, fc2p, 544, 544, 512, 544 * 512);
  pack_w<<<nb(512LL * 512, TB), TB, 0, stream>>>(fc3w, fc3p, 512, 512, 512, 512 * 512);

  // --- sequence model ---
  build_x0<<<nb((long long)Mp * 128, TB), TB, 0, stream>>>(x, m2, auxb, X0, Mp);

  dim3 gN512(512 / 128, Mp / 64);
  dim3 gN1536(1536 / 128, Mp / 64);

  gemm_bf16<<<gN512, 256, 0, stream>>>(X0, Iwp, I_b, xi, Mp, 128, 512, 0);            // xi
  gemm_bf16<<<gN1536, 256, 0, stream>>>(xi, wih1p, bih1, xp, Mp, 512, 1536, 0);       // xp1
  gru_scan<<<1, 1024, 0, stream>>>(xp, whh1p, bhh1, g1, T);                           // GRU1
  build_cat<<<nb((long long)Mp * 544, TB), TB, 0, stream>>>(g1, 512, xi, 512, auxb, 32, X2, Mp);
  gemm_bf16<<<gN1536, 256, 0, stream>>>(X2, wih2p, bih2, xp, Mp, 544, 1536, 0);       // xp2
  gru_scan<<<1, 1024, 0, stream>>>(xp, whh2p, bhh2, xi, T);                           // GRU2 -> xi
  build_cat<<<nb((long long)Mp * 544, TB), TB, 0, stream>>>(xi, 512, X2, 544, auxb, 64, X3, Mp);
  gemm_bf16<<<gN512, 256, 0, stream>>>(X3, fc1p, fc1b, g1, Mp, 544, 512, 1);          // f1
  build_cat<<<nb((long long)Mp * 544, TB), TB, 0, stream>>>(g1, 512, nullptr, 0, auxb, 96, X2, Mp);
  gemm_bf16<<<gN512, 256, 0, stream>>>(X2, fc2p, fc2b, xi, Mp, 544, 512, 1);          // f2
  gemm_bf16<<<gN512, 256, 0, stream>>>(xi, fc3p, fc3b, (float*)d_out, Mreal, 512, 512, 0);
}
